// HyPlantSensorSimulator_20813411517107
// MI455X (gfx1250) — matrix operand order repaired and checked by reading, not compiled, not run on hardware
//
#include <hip/hip_runtime.h>
#include <hip/hip_bf16.h>
#include <stdint.h>

// ---------------- problem constants ----------------
constexpr int NHR   = 3637;        // high-res spectral length
constexpr int LBLUR = 3638;        // blur output length (C+1)
constexpr int KTAP  = 694;         // gaussian taps
constexpr int NSEN  = 200;         // sensor bands
constexpr int BSZ = 2, HSZ = 64, WSZ = 64;
constexpr float HIGH_RES = 0.0055f;

// ---------------- GEMM mapping ----------------
constexpr int KPAD  = 736;         // K taps padded: 694 + 15 window + pad -> 23*32
constexpr int NCH   = KPAD / 32;   // 23 K-chunks of 32 (bf16 WMMA K)
constexpr int JT    = 228;         // ceil(3638/16) spectral 16-tiles
constexpr int TILEP = 16;          // pixels per workgroup
constexpr int NTILES = (BSZ * HSZ * WSZ) / TILEP;  // 512

// ---------------- LDS layout (bytes) ----------------
// spec row stride: 4488 bf16 = 2244 dwords; 2244 % 64 == 4 -> 16 rows of an
// A-fragment b128 hit disjoint bank quads. Row needs >= 347 + 3632-347+736.
constexpr int SPEC_STRIDE = 4488;                       // in bf16 elements
constexpr int SPEC_BYTES  = TILEP * SPEC_STRIDE * 2;    // 143,616
// blur row stride: 3720 bf16 = 1860 dwords; 1860 % 64 == 4.
constexpr int BLUR_STRIDE = 3720;                       // in bf16 elements
constexpr int BLUR_BYTES  = TILEP * BLUR_STRIDE * 2;    // 119,040
constexpr int TOEP_BYTES  = NCH * 32 * 32;              // 23,552 (per chunk: 32 lanes * 32B)
constexpr int LDS_BYTES   = SPEC_BYTES + BLUR_BYTES + TOEP_BYTES;  // 286,208 < 320K

typedef __attribute__((ext_vector_type(8)))  float  v8f;
typedef __attribute__((ext_vector_type(16))) __bf16 v16bf;

union ABfrag { uint4 u[2]; v16bf v; };

static __device__ __forceinline__ unsigned short f2bf(float f) {
    unsigned int u = __float_as_uint(f);
    unsigned int r = (u + 0x7FFFu + ((u >> 16) & 1u)) >> 16;   // RNE
    return (unsigned short)r;
}
static __device__ __forceinline__ float bf2f(unsigned short s) {
    return __uint_as_float(((unsigned int)s) << 16);
}

// ============================================================
// Setup: build normalized Gaussian + Toeplitz B fragments (bf16,
// pre-packed per K-chunk in B lane order k = 16*(lane>>4) + s, same order as A).
//   T[q][n] = kernel[q - n]  (0 if q-n outside [0, K-1])
//   toep[c][lane][j] = pair( T[c*32 + 16*(lane>>4) + 2j    ][lane&15],
//                            T[c*32 + 16*(lane>>4) + 2j + 1][lane&15] )
// ============================================================
__global__ __launch_bounds__(256) void hyplant_setup_kernel(
    const float* __restrict__ delta_sigma,
    unsigned short* __restrict__ toep)
{
    __shared__ float kcache[KPAD];
    __shared__ float sinv;
    const int t = threadIdx.x;

    const float sigma = (0.27f + delta_sigma[0]) * 2.3548f;
    const float inv   = 1.0f / sigma;
    for (int i = t; i < KPAD; i += 256) {
        float v = 0.0f;
        if (i < KTAP) {
            float x = -3.0f * sigma + HIGH_RES * (float)i;
            float z = x * inv;
            v = __expf(-0.5f * z * z);
        }
        kcache[i] = v;
    }
    __syncthreads();
    if (t == 0) {
        float s = 0.0f;
        for (int i = 0; i < KTAP; ++i) s += kcache[i];
        sinv = 1.0f / s;
    }
    __syncthreads();
    const float sc = sinv;

    // 23 chunks * 32 lanes * 8 pairs
    for (int e = t; e < NCH * 32 * 8; e += 256) {
        const int c    = e >> 8;          // /256
        const int r    = e & 255;
        const int lane = r >> 3;
        const int j    = r & 7;
        const int n     = lane & 15;
        const int khalf = (lane >> 4) << 4;       // 0 or 16
        const int q     = c * 32 + khalf + 2 * j; // K index of low half
        const int k0 = q - n;
        const int k1 = q + 1 - n;
        const float v0 = (k0 >= 0 && k0 < KTAP) ? kcache[k0] * sc : 0.0f;
        const float v1 = (k1 >= 0 && k1 < KTAP) ? kcache[k1] * sc : 0.0f;
        const int base = c * 512 + lane * 16 + j * 2;   // ushort index
        toep[base]     = f2bf(v0);
        toep[base + 1] = f2bf(v1);
    }
}

// ============================================================
// Main: per 16-pixel tile — stream spec to LDS (f32->bf16),
// Toeplitz-GEMM blur via v_wmma_f32_16x16x32_bf16, then per-pixel
// shifted linear resample from LDS to the [B,200,H,W] output.
// ============================================================
__global__ __launch_bounds__(256, 1) void hyplant_blur_sample_kernel(
    const float* __restrict__ Lin,            // [B, NHR, H, W]
    const float* __restrict__ dlam,           // [B, H, W]
    const float* __restrict__ swl,            // [200]
    const unsigned short* __restrict__ toepG, // [NCH*32*16] bf16 pairs
    float* __restrict__ out)                  // [B, 200, H, W]
{
    extern __shared__ __align__(16) char smem[];
    unsigned short* spec = (unsigned short*)smem;
    unsigned short* blur = (unsigned short*)(smem + SPEC_BYTES);
    unsigned short* toep = (unsigned short*)(smem + SPEC_BYTES + BLUR_BYTES);

    const int t    = threadIdx.x;
    const int lane = t & 31;
    const int wv   = t >> 5;

    const int pbase = blockIdx.x * TILEP;     // flat pixel base in B*H*W
    const int b     = pbase >> 12;            // /(H*W)
    const int hw    = pbase & 4095;
    const int hrow  = hw >> 6;
    const int w0    = hw & 63;

    // ---- zero spec halo + copy Toeplitz to LDS ----
    {
        uint4 z; z.x = z.y = z.z = z.w = 0u;
        uint4* s4 = (uint4*)smem;
        for (int i = t; i < SPEC_BYTES / 16; i += 256) s4[i] = z;
        const unsigned int* src = (const unsigned int*)toepG;
        unsigned int* dst = (unsigned int*)toep;
        for (int i = t; i < TOEP_BYTES / 4; i += 256) dst[i] = src[i];
    }
    __syncthreads();

    // ---- load 16 pixels' spectra, convert to bf16 into LDS ----
    {
        const int p = t & 15;
        const float* src = Lin + (size_t)b * NHR * 4096 + (size_t)(hrow * 64 + w0 + p);
        unsigned short* dstrow = spec + p * SPEC_STRIDE;
        for (int c = t >> 4; c < NHR; c += 16) {        // 16 lanes read 64B rows
            dstrow[347 + c] = f2bf(src[(size_t)c * 4096]);
        }
    }
    __syncthreads();

    // ---- Toeplitz GEMM: blur[16 x 3648] = spec x T, bf16 WMMA / f32 acc ----
    const int ksA  = (lane >> 4) << 4;                  // A K-half: 0 or 16
    const int prow = lane & 15;
    const char* aRow  = (const char*)(spec + prow * SPEC_STRIDE);
    const char* bBase = (const char*)toep + lane * 32;

    for (int jt = wv; jt < JT; jt += 8) {
        const int j0 = jt * 16;
        v8f acc = {};
        const char* ap = aRow + (size_t)(j0 + ksA) * 2; // 16B aligned (j0+ks mult of 8)
        const char* bp = bBase;
        for (int c = 0; c < NCH; ++c) {
            ABfrag a, bm;
            a.u[0]  = *(const uint4*)(ap);              // K = ks .. ks+7
            a.u[1]  = *(const uint4*)(ap + 16);         // K = ks+8 .. ks+15
            bm.u[0] = *(const uint4*)(bp);
            bm.u[1] = *(const uint4*)(bp + 16);
            acc = __builtin_amdgcn_wmma_f32_16x16x32_bf16(
                false, a.v, false, bm.v, (short)0, acc, false, false);
            ap += 64;                                   // advance 32 K-taps (bf16)
            bp += 1024;                                 // next Toeplitz chunk
        }
        // D layout: lane -> column n, VGPR r -> row r + 8*(lane>>4)
        const int n  = lane & 15;
        const int m0 = (lane >> 4) << 3;
        #pragma unroll
        for (int r = 0; r < 8; ++r)
            blur[(m0 + r) * BLUR_STRIDE + j0 + n] = f2bf(acc[r]);
    }
    __syncthreads();

    // ---- per-pixel shifted linear resample (zeros outside [0, L-1]) ----
    {
        const int p = t & 15;
        const float dlp = dlam[pbase + p];
        const unsigned short* brow = blur + p * BLUR_STRIDE;
        for (int s = t >> 4; s < NSEN; s += 16) {
            const float lam = swl[s] + dlp;
            const float idx = (lam - 750.0f) * ((float)LBLUR / 20.0f) - 0.5f;
            const float f0  = floorf(idx);
            const int   i0  = (int)f0;
            const float w1  = idx - f0;
            int ic0 = i0     < 0 ? 0 : (i0     > LBLUR - 1 ? LBLUR - 1 : i0);
            int ic1 = i0 + 1 < 0 ? 0 : (i0 + 1 > LBLUR - 1 ? LBLUR - 1 : i0 + 1);
            float g0 = bf2f(brow[ic0]);
            float g1 = bf2f(brow[ic1]);
            if (i0     < 0 || i0     > LBLUR - 1) g0 = 0.0f;
            if (i0 + 1 < 0 || i0 + 1 > LBLUR - 1) g1 = 0.0f;
            const float o = g0 * (1.0f - w1) + g1 * w1;
            out[(((size_t)b * NSEN + s) * 64 + hrow) * 64 + (w0 + p)] = o;
        }
    }
}

// ============================================================
extern "C" void kernel_launch(void* const* d_in, const int* in_sizes, int n_in,
                              void* d_out, int out_size, void* d_ws, size_t ws_size,
                              hipStream_t stream) {
    (void)in_sizes; (void)n_in; (void)out_size; (void)ws_size;
    const float* Lin  = (const float*)d_in[0];   // [B, NHR, H, W] f32
    const float* dlam = (const float*)d_in[1];   // [B, H, W] f32
    const float* dsig = (const float*)d_in[2];   // [1] f32
    const float* swl  = (const float*)d_in[3];   // [200] f32
    float* out = (float*)d_out;

    unsigned short* toep = (unsigned short*)d_ws;   // 23,552 B of scratch

    hipLaunchKernelGGL(hyplant_setup_kernel, dim3(1), dim3(256), 0, stream,
                       dsig, toep);
    hipLaunchKernelGGL(hyplant_blur_sample_kernel, dim3(NTILES), dim3(256),
                       LDS_BYTES, stream, Lin, dlam, swl, toep, out);
}
